// RWKV_Block_798863917075
// MI455X (gfx1250) — matrix operand order repaired and checked by reading, not compiled, not run on hardware
//
#include <hip/hip_runtime.h>
#include <cstdint>
#include <cstddef>

// ---------- types ----------
typedef _Float16 f16_t;
typedef __attribute__((ext_vector_type(16))) _Float16 v16h;
typedef __attribute__((ext_vector_type(8)))  _Float16 v8h;
typedef __attribute__((ext_vector_type(8)))  float    v8f;

#define LIF_TAU 2.0f
#define LIF_VTH 1.0f

__device__ __forceinline__ float dev_sigmoid(float x) {
    return 1.0f / (1.0f + __expf(-x));
}

// forward of atan_spike is a Heaviside; LIF with TAU=2 over 4 identical inputs
__device__ __forceinline__ float lif_mean4(float x) {
    float v = 0.0f, s = 0.0f;
#pragma unroll
    for (int i = 0; i < 4; ++i) {
        v = 0.5f * (v + x);               // v + (x - v)/TAU, TAU=2
        float sp = ((v - LIF_VTH) >= 0.0f) ? 1.0f : 0.0f;
        s += sp;
        v *= (1.0f - sp);
    }
    return 0.25f * s;
}

// ---------- elementwise kernels ----------
__global__ void k_build_xc(const float* __restrict__ x, const float* __restrict__ rt,
                           float* __restrict__ xc, int B, int P1, int C) {
    size_t T = 2 * (size_t)P1;
    size_t n = (size_t)B * T * C;
    size_t i = (size_t)blockIdx.x * blockDim.x + threadIdx.x;
    if (i >= n) return;
    int c = (int)(i % C);
    size_t bt = i / C;
    int t = (int)(bt % T);
    int b = (int)(bt / T);
    float v = (t < P1) ? x[((size_t)b * P1 + t) * C + c]
                       : rt[((size_t)(t - P1)) * C + c];
    xc[i] = v;
}

__global__ __launch_bounds__(256)
void k_layernorm(const float* __restrict__ in, float* __restrict__ out, int C) {
    __shared__ float s1[256];
    __shared__ float s2[256];
    size_t row = blockIdx.x;
    int tid = threadIdx.x;
    const float* p = in + row * (size_t)C;
    float a = 0.0f, b = 0.0f;
    for (int c = tid; c < C; c += 256) { float v = p[c]; a += v; b += v * v; }
    s1[tid] = a; s2[tid] = b;
    __syncthreads();
    for (int off = 128; off > 0; off >>= 1) {
        if (tid < off) { s1[tid] += s1[tid + off]; s2[tid] += s2[tid + off]; }
        __syncthreads();
    }
    float m = s1[0] / (float)C;
    float var = s2[0] / (float)C - m * m;
    float rstd = rsqrtf(var + 1e-6f);
    float* q = out + row * (size_t)C;
    for (int c = tid; c < C; c += 256) q[c] = (p[c] - m) * rstd;
}

// time-shift + 3 mixes -> f16 GEMM inputs
__global__ void k_tshift_mix3(const float* __restrict__ ln,
                              const float* __restrict__ tk, const float* __restrict__ tv,
                              const float* __restrict__ tr,
                              f16_t* __restrict__ ok, f16_t* __restrict__ ov, f16_t* __restrict__ orr,
                              int T, int C, size_t n) {
    size_t i = (size_t)blockIdx.x * blockDim.x + threadIdx.x;
    if (i >= n) return;
    int c = (int)(i % C);
    int t = (int)((i / C) % T);
    float h = ln[i];
    float xx = (t == 0) ? 0.0f : ln[i - C];
    float mk = tk[c], mv = tv[c], mr = tr[c];
    ok[i]  = (f16_t)(h * mk + xx * (1.0f - mk));
    ov[i]  = (f16_t)(h * mv + xx * (1.0f - mv));
    orr[i] = (f16_t)(h * mr + xx * (1.0f - mr));
}

__global__ void k_tshift_mix2(const float* __restrict__ ln,
                              const float* __restrict__ tk, const float* __restrict__ tr,
                              f16_t* __restrict__ ok, f16_t* __restrict__ orr,
                              int T, int C, size_t n) {
    size_t i = (size_t)blockIdx.x * blockDim.x + threadIdx.x;
    if (i >= n) return;
    int c = (int)(i % C);
    int t = (int)((i / C) % T);
    float h = ln[i];
    float xx = (t == 0) ? 0.0f : ln[i - C];
    float mk = tk[c], mr = tr[c];
    ok[i]  = (f16_t)(h * mk + xx * (1.0f - mk));
    orr[i] = (f16_t)(h * mr + xx * (1.0f - mr));
}

__global__ void k_f32_to_f16(const float* __restrict__ in, f16_t* __restrict__ out, size_t n) {
    size_t i = (size_t)blockIdx.x * blockDim.x + threadIdx.x;
    if (i < n) out[i] = (f16_t)in[i];
}

// ---------- WMMA GEMM:  Y[m,n] = sum_k X[m,k] * W[n,k]  (both K-major) ----------
// Wave tile 32x64 (2 M-frags x 4 N-frags = 8 WMMAs per k-step).
// 8 waves arranged 4(M) x 2(N) -> block tile 128x128.
// Register double-buffered: k-step i+1 fragment loads are issued before the
// 8 WMMAs of k-step i, so loads fly under the matrix ops.
// MODE: 0 = plain f32, 1 = sigmoid f32, 2 = relu^2 -> f16, 3 = +bias[n] f32
template <int MODE>
__global__ __launch_bounds__(256)
void k_gemm_wmma(const f16_t* __restrict__ X, const f16_t* __restrict__ W,
                 float* __restrict__ outF, f16_t* __restrict__ outH,
                 const float* __restrict__ bias, int M, int N, int K) {
    int tid  = threadIdx.x;
    int wave = tid >> 5;
    int lane = tid & 31;
    int h    = lane >> 4;      // wave half
    int l16  = lane & 15;
    int waveM = wave >> 1;     // 0..3
    int waveN = wave & 1;      // 0..1
    int m0 = blockIdx.y * 128 + waveM * 32;
    int n0 = blockIdx.x * 128 + waveN * 64;

    // A fragment base pointers (rows m0+l16 and m0+16+l16), K-offset h*16 pattern
    const f16_t* xr0 = X + (size_t)(m0 + l16) * K + h * 16;
    const f16_t* xr1 = X + (size_t)(m0 + 16 + l16) * K + h * 16;
    // B fragment base pointers (col n, contiguous 16 halves at k0 + h*16)
    const f16_t* wp0 = W + (size_t)(n0 + 0 * 16 + l16) * K + h * 16;
    const f16_t* wp1 = W + (size_t)(n0 + 1 * 16 + l16) * K + h * 16;
    const f16_t* wp2 = W + (size_t)(n0 + 2 * 16 + l16) * K + h * 16;
    const f16_t* wp3 = W + (size_t)(n0 + 3 * 16 + l16) * K + h * 16;

    v8f acc[2][4];
#pragma unroll
    for (int mi = 0; mi < 2; ++mi)
#pragma unroll
        for (int j = 0; j < 4; ++j)
#pragma unroll
            for (int e = 0; e < 8; ++e) acc[mi][j][e] = 0.0f;

#define LOAD_A(ptr, k0, dst)                                              \
    do {                                                                  \
        v8h lo_ = *reinterpret_cast<const v8h*>((ptr) + (k0));            \
        v8h hi_ = *reinterpret_cast<const v8h*>((ptr) + (k0) + 8);        \
        _Pragma("unroll")                                                 \
        for (int e_ = 0; e_ < 8; ++e_) { dst[e_] = lo_[e_]; dst[e_ + 8] = hi_[e_]; } \
    } while (0)

    v16h a0, a1, b0, b1, b2, b3;
    LOAD_A(xr0, 0, a0);
    LOAD_A(xr1, 0, a1);
    b0 = *reinterpret_cast<const v16h*>(wp0);
    b1 = *reinterpret_cast<const v16h*>(wp1);
    b2 = *reinterpret_cast<const v16h*>(wp2);
    b3 = *reinterpret_cast<const v16h*>(wp3);

    // steady state: prefetch next k-step fragments, then 8 WMMAs on current
    for (int k0 = 0; k0 < K - 32; k0 += 32) {
        int kn = k0 + 32;
        v16h na0, na1, nb0, nb1, nb2, nb3;
        LOAD_A(xr0, kn, na0);
        LOAD_A(xr1, kn, na1);
        nb0 = *reinterpret_cast<const v16h*>(wp0 + kn);
        nb1 = *reinterpret_cast<const v16h*>(wp1 + kn);
        nb2 = *reinterpret_cast<const v16h*>(wp2 + kn);
        nb3 = *reinterpret_cast<const v16h*>(wp3 + kn);
        __builtin_prefetch(xr0 + kn + 32, 0, 1);

        acc[0][0] = __builtin_amdgcn_wmma_f32_16x16x32_f16(false, a0, false, b0, (short)0, acc[0][0], false, false);
        acc[0][1] = __builtin_amdgcn_wmma_f32_16x16x32_f16(false, a0, false, b1, (short)0, acc[0][1], false, false);
        acc[0][2] = __builtin_amdgcn_wmma_f32_16x16x32_f16(false, a0, false, b2, (short)0, acc[0][2], false, false);
        acc[0][3] = __builtin_amdgcn_wmma_f32_16x16x32_f16(false, a0, false, b3, (short)0, acc[0][3], false, false);
        acc[1][0] = __builtin_amdgcn_wmma_f32_16x16x32_f16(false, a1, false, b0, (short)0, acc[1][0], false, false);
        acc[1][1] = __builtin_amdgcn_wmma_f32_16x16x32_f16(false, a1, false, b1, (short)0, acc[1][1], false, false);
        acc[1][2] = __builtin_amdgcn_wmma_f32_16x16x32_f16(false, a1, false, b2, (short)0, acc[1][2], false, false);
        acc[1][3] = __builtin_amdgcn_wmma_f32_16x16x32_f16(false, a1, false, b3, (short)0, acc[1][3], false, false);

        a0 = na0; a1 = na1; b0 = nb0; b1 = nb1; b2 = nb2; b3 = nb3;
    }
    // peeled final k-step
    acc[0][0] = __builtin_amdgcn_wmma_f32_16x16x32_f16(false, a0, false, b0, (short)0, acc[0][0], false, false);
    acc[0][1] = __builtin_amdgcn_wmma_f32_16x16x32_f16(false, a0, false, b1, (short)0, acc[0][1], false, false);
    acc[0][2] = __builtin_amdgcn_wmma_f32_16x16x32_f16(false, a0, false, b2, (short)0, acc[0][2], false, false);
    acc[0][3] = __builtin_amdgcn_wmma_f32_16x16x32_f16(false, a0, false, b3, (short)0, acc[0][3], false, false);
    acc[1][0] = __builtin_amdgcn_wmma_f32_16x16x32_f16(false, a1, false, b0, (short)0, acc[1][0], false, false);
    acc[1][1] = __builtin_amdgcn_wmma_f32_16x16x32_f16(false, a1, false, b1, (short)0, acc[1][1], false, false);
    acc[1][2] = __builtin_amdgcn_wmma_f32_16x16x32_f16(false, a1, false, b2, (short)0, acc[1][2], false, false);
    acc[1][3] = __builtin_amdgcn_wmma_f32_16x16x32_f16(false, a1, false, b3, (short)0, acc[1][3], false, false);
#undef LOAD_A

    // epilogue: D vgpr r -> row (r | r+8 by wave half), col n0 + j*16 + l16
#pragma unroll
    for (int mi = 0; mi < 2; ++mi) {
#pragma unroll
        for (int j = 0; j < 4; ++j) {
            int n = n0 + j * 16 + l16;
            float bv = (MODE == 3) ? bias[n] : 0.0f;
#pragma unroll
            for (int r = 0; r < 8; ++r) {
                int m = m0 + mi * 16 + r + h * 8;
                float val = acc[mi][j][r];
                if (MODE == 1) val = dev_sigmoid(val);
                else if (MODE == 2) { val = val > 0.0f ? val : 0.0f; val = val * val; }
                else if (MODE == 3) val += bv;
                size_t o = (size_t)m * N + n;
                if (MODE == 2) outH[o] = (f16_t)val;
                else           outF[o] = val;
            }
        }
    }
}

// ---------- serial wkv scan fused with sigmoid(r) * y -> f16 ----------
__global__ void k_wkv_sry(const float* __restrict__ td, const float* __restrict__ tf,
                          const float* __restrict__ kf, const float* __restrict__ vf,
                          const float* __restrict__ rf, f16_t* __restrict__ sry,
                          int B, int T, int C) {
    int i = blockIdx.x * blockDim.x + threadIdx.x;
    if (i >= B * C) return;
    int c = i % C;
    int b = i / C;
    float w = -__expf(td[c]);
    float u = tf[c];
    float aa = 0.0f, bb = 0.0f, pp = -1e38f;
    size_t base = (size_t)b * T * C + c;
    for (int t = 0; t < T; ++t) {
        size_t o = base + (size_t)t * C;
        float kt = kf[o], vt = vf[o];
        float ww = u + kt;
        float p  = fmaxf(pp, ww);
        float e1 = __expf(pp - p), e2 = __expf(ww - p);
        float y  = (e1 * aa + e2 * vt) / (e1 * bb + e2);
        float ww2 = pp + w;
        float p2  = fmaxf(ww2, kt);
        float f1 = __expf(ww2 - p2), f2 = __expf(kt - p2);
        aa = f1 * aa + f2 * vt;
        bb = f1 * bb + f2;
        pp = p2;
        float sr = dev_sigmoid(rf[o]);
        sry[o] = (f16_t)(sr * y);
    }
}

// xc += lif_mean4( add * (mul ? mul : 1) ), in place
__global__ void k_lif_residual(float* __restrict__ xc, const float* __restrict__ add,
                               const float* __restrict__ mul, size_t n) {
    size_t i = (size_t)blockIdx.x * blockDim.x + threadIdx.x;
    if (i >= n) return;
    float x = add[i];
    if (mul) x *= mul[i];
    xc[i] = xc[i] + lif_mean4(x);
}

// rtok16[(b*P1+p)*C+c] = f16( xc[(b*2P1 + P1 + p)*C + c] )
__global__ void k_rtok_f16(const float* __restrict__ xc, f16_t* __restrict__ out,
                           int B, int P1, int C) {
    size_t n = (size_t)B * P1 * C;
    size_t i = (size_t)blockIdx.x * blockDim.x + threadIdx.x;
    if (i >= n) return;
    int c = (int)(i % C);
    size_t bp = i / C;
    int p = (int)(bp % P1);
    int b = (int)(bp / P1);
    out[i] = (f16_t)xc[((size_t)b * 2 * P1 + P1 + p) * C + c];
}

// x2[(b*P+p)*P + q] = f16( r1[(b*P+q)*P + p] )   (per-batch transpose)
__global__ void k_transpose_cvt(const float* __restrict__ r1, f16_t* __restrict__ x2,
                                int B, int P) {
    size_t n = (size_t)B * P * P;
    size_t i = (size_t)blockIdx.x * blockDim.x + threadIdx.x;
    if (i >= n) return;
    int q = (int)(i % P);
    size_t bp = i / P;
    int p = (int)(bp % P);
    int b = (int)(bp / P);
    x2[i] = (f16_t)r1[((size_t)b * P + q) * P + p];
}

// out[b,p,c] = xo[b,p,c] * (1 + r2[b,p,c]);   xo = xc[:, :P1]
__global__ void k_final(const float* __restrict__ xc, const float* __restrict__ r2,
                        float* __restrict__ out, int B, int P1, int C) {
    size_t n = (size_t)B * P1 * C;
    size_t i = (size_t)blockIdx.x * blockDim.x + threadIdx.x;
    if (i >= n) return;
    int c = (int)(i % C);
    size_t bp = i / C;
    int p = (int)(bp % P1);
    int b = (int)(bp / P1);
    float xo = xc[((size_t)b * 2 * P1 + p) * C + c];
    out[i] = xo * (1.0f + r2[i]);
}

// ---------- host ----------
static inline size_t ceil_div(size_t a, size_t b) { return (a + b - 1) / b; }

extern "C" void kernel_launch(void* const* d_in, const int* in_sizes, int n_in,
                              void* d_out, int out_size, void* d_ws, size_t ws_size,
                              hipStream_t stream) {
    (void)in_sizes; (void)n_in; (void)out_size; (void)ws_size;
    const int B = 8, P1 = 256, C = 1024, H = 4096, T = 2 * P1;
    const size_t NTC = (size_t)B * T * C;          // 4M
    const size_t NPC = (size_t)B * P1 * C;         // 2M

    const float* x   = (const float*)d_in[0];
    const float* rt  = (const float*)d_in[1];
    const float* td  = (const float*)d_in[2];
    const float* tf  = (const float*)d_in[3];
    const float* tmk = (const float*)d_in[4];
    const float* tmv = (const float*)d_in[5];
    const float* tmr = (const float*)d_in[6];
    const float* Wk  = (const float*)d_in[7];
    const float* Wv  = (const float*)d_in[8];
    const float* Wr  = (const float*)d_in[9];
    const float* Wo  = (const float*)d_in[10];
    const float* cmk = (const float*)d_in[11];
    const float* cmr = (const float*)d_in[12];
    const float* Wck = (const float*)d_in[13];
    const float* Wcv = (const float*)d_in[14];
    const float* Wcr = (const float*)d_in[15];
    const float* Wrm = (const float*)d_in[16];
    const float* brm = (const float*)d_in[17];
    const float* Wam = (const float*)d_in[18];
    const float* bam = (const float*)d_in[19];
    float* out = (float*)d_out;

    // workspace bump allocator (256B aligned)
    uintptr_t cur = (uintptr_t)d_ws;
    auto alloc = [&](size_t bytes) -> void* {
        uintptr_t p = (cur + 255) & ~(uintptr_t)255;
        cur = p + bytes;
        return (void*)p;
    };

    float* xcb   = (float*)alloc(NTC * 4);             // xc -> xc2 -> xc3 (in place)
    float* lnb   = (float*)alloc(NTC * 4);             // ln1 then ln2
    f16_t* xk16  = (f16_t*)alloc(NTC * 2);             // xk -> xk2 -> rtok16
    f16_t* xv16  = (f16_t*)alloc(NTC * 2);             // xv -> sry16
    f16_t* xr16  = (f16_t*)alloc(NTC * 2);             // xr -> xr2 -> x2t16
    float* kf    = (float*)alloc(NTC * 4);             // k -> att -> r2
    float* vf    = (float*)alloc(NTC * 4);             // v -> kv
    float* rf    = (float*)alloc(NTC * 4);             // r -> sig2
    f16_t* wk16  = (f16_t*)alloc((size_t)C * C * 2);
    f16_t* wv16  = (f16_t*)alloc((size_t)C * C * 2);
    f16_t* wr16  = (f16_t*)alloc((size_t)C * C * 2);
    f16_t* wo16  = (f16_t*)alloc((size_t)C * C * 2);
    f16_t* wck16 = (f16_t*)alloc((size_t)H * C * 2);
    f16_t* wcv16 = (f16_t*)alloc((size_t)C * H * 2);
    f16_t* wcr16 = (f16_t*)alloc((size_t)C * C * 2);
    f16_t* wrm16 = (f16_t*)alloc((size_t)P1 * C * 2);
    f16_t* wam16 = (f16_t*)alloc((size_t)C * P1 * 2);
    f16_t* kbuf16 = (f16_t*)alloc((size_t)B * T * H * 2);   // relu^2(k) f16
    float* r1    = (float*)alloc((size_t)B * P1 * P1 * 4);

    // aliases
    f16_t* sry16 = xv16;
    float* att   = kf;
    float* kvb   = vf;
    float* sig2  = rf;
    f16_t* rtok16 = xk16;
    f16_t* x2t16  = xr16;
    float* r2    = kf;

    const int BT = 256;
    dim3 blk(BT);

    // 1. xc = concat(x, rt)
    k_build_xc<<<dim3((unsigned)ceil_div(NTC, BT)), blk, 0, stream>>>(x, rt, xcb, B, P1, C);
    // 2. ln1
    k_layernorm<<<dim3((unsigned)(B * T)), blk, 0, stream>>>(xcb, lnb, C);
    // 3. time-shift mixes -> f16
    k_tshift_mix3<<<dim3((unsigned)ceil_div(NTC, BT)), blk, 0, stream>>>(
        lnb, tmk, tmv, tmr, xk16, xv16, xr16, T, C, NTC);
    // 4. weight conversions f32 -> f16
    {
        size_t cc = (size_t)C * C, hc = (size_t)H * C, pc = (size_t)P1 * C;
        k_f32_to_f16<<<dim3((unsigned)ceil_div(cc, BT)), blk, 0, stream>>>(Wk, wk16, cc);
        k_f32_to_f16<<<dim3((unsigned)ceil_div(cc, BT)), blk, 0, stream>>>(Wv, wv16, cc);
        k_f32_to_f16<<<dim3((unsigned)ceil_div(cc, BT)), blk, 0, stream>>>(Wr, wr16, cc);
        k_f32_to_f16<<<dim3((unsigned)ceil_div(cc, BT)), blk, 0, stream>>>(Wo, wo16, cc);
        k_f32_to_f16<<<dim3((unsigned)ceil_div(hc, BT)), blk, 0, stream>>>(Wck, wck16, hc);
        k_f32_to_f16<<<dim3((unsigned)ceil_div(hc, BT)), blk, 0, stream>>>(Wcv, wcv16, hc);
        k_f32_to_f16<<<dim3((unsigned)ceil_div(cc, BT)), blk, 0, stream>>>(Wcr, wcr16, cc);
        k_f32_to_f16<<<dim3((unsigned)ceil_div(pc, BT)), blk, 0, stream>>>(Wrm, wrm16, pc);
        k_f32_to_f16<<<dim3((unsigned)ceil_div(pc, BT)), blk, 0, stream>>>(Wam, wam16, pc);
    }
    // 5-7. k, v, r GEMMs  (M=4096, N=1024, K=1024), block tile 128x128
    {
        dim3 g(C / 128, (B * T) / 128);
        k_gemm_wmma<0><<<g, blk, 0, stream>>>(xk16, wk16, kf, nullptr, nullptr, B * T, C, C);
        k_gemm_wmma<0><<<g, blk, 0, stream>>>(xv16, wv16, vf, nullptr, nullptr, B * T, C, C);
        k_gemm_wmma<0><<<g, blk, 0, stream>>>(xr16, wr16, rf, nullptr, nullptr, B * T, C, C);
    }
    // 8. wkv scan fused with sigmoid(r)*y -> f16
    k_wkv_sry<<<dim3((unsigned)ceil_div((size_t)B * C, BT)), blk, 0, stream>>>(
        td, tf, kf, vf, rf, sry16, B, T, C);
    // 9. att = (sr*y) @ Wo^T
    k_gemm_wmma<0><<<dim3(C / 128, (B * T) / 128), blk, 0, stream>>>(
        sry16, wo16, att, nullptr, nullptr, B * T, C, C);
    // 10. xc += LIF(att)
    k_lif_residual<<<dim3((unsigned)ceil_div(NTC, BT)), blk, 0, stream>>>(xcb, att, nullptr, NTC);
    // 11. ln2
    k_layernorm<<<dim3((unsigned)(B * T)), blk, 0, stream>>>(xcb, lnb, C);
    // 12. channel-mix mixes -> f16
    k_tshift_mix2<<<dim3((unsigned)ceil_div(NTC, BT)), blk, 0, stream>>>(
        lnb, cmk, cmr, xk16, xr16, T, C, NTC);
    // 13. relu^2( xk2 @ Wck^T ) -> f16   (M=4096, N=4096, K=1024)
    k_gemm_wmma<2><<<dim3(H / 128, (B * T) / 128), blk, 0, stream>>>(
        xk16, wck16, nullptr, kbuf16, nullptr, B * T, H, C);
    // 14. kv = kbuf @ Wcv^T              (M=4096, N=1024, K=4096)
    k_gemm_wmma<0><<<dim3(C / 128, (B * T) / 128), blk, 0, stream>>>(
        kbuf16, wcv16, kvb, nullptr, nullptr, B * T, C, H);
    // 15. sig2 = sigmoid( xr2 @ Wcr^T )
    k_gemm_wmma<1><<<dim3(C / 128, (B * T) / 128), blk, 0, stream>>>(
        xr16, wcr16, sig2, nullptr, nullptr, B * T, C, C);
    // 16. xc += LIF(sig2 * kv)
    k_lif_residual<<<dim3((unsigned)ceil_div(NTC, BT)), blk, 0, stream>>>(xcb, kvb, sig2, NTC);
    // 17. rtok -> f16
    k_rtok_f16<<<dim3((unsigned)ceil_div(NPC, BT)), blk, 0, stream>>>(xcb, rtok16, B, P1, C);
    // 18. r1 = rtok @ Wrm^T + brm        (M=2048, N=256, K=1024)
    k_gemm_wmma<3><<<dim3(P1 / 128, (B * P1) / 128), blk, 0, stream>>>(
        rtok16, wrm16, r1, nullptr, brm, B * P1, P1, C);
    // 19. per-batch transpose -> f16
    k_transpose_cvt<<<dim3((unsigned)ceil_div((size_t)B * P1 * P1, BT)), blk, 0, stream>>>(
        r1, x2t16, B, P1);
    // 20. r2 = x2t @ Wam^T + bam         (M=2048, N=1024, K=256)
    k_gemm_wmma<3><<<dim3(C / 128, (B * P1) / 128), blk, 0, stream>>>(
        x2t16, wam16, r2, nullptr, bam, B * P1, C, P1);
    // 21. out = xo * (1 + r2)
    k_final<<<dim3((unsigned)ceil_div(NPC, BT)), blk, 0, stream>>>(xcb, r2, out, B, P1, C);
}
